// LiquidNeuralNetwork_25666724561199
// MI455X (gfx1250) — matrix operand order repaired and checked by reading, not compiled, not run on hardware
//
#include <hip/hip_runtime.h>
#include <hip/hip_bf16.h>

// ---------------------------------------------------------------------------
// LiquidNeuralNetwork forward for MI455X (gfx1250), wave32 + WMMA f16->f32.
// GEMM B-tiles are staged global->LDS with CDNA5 async-tensor copies
// (global_load_async_to_lds_b128 / ASYNCcnt), then fed to
// v_wmma_f32_16x16x32_f16.
// ---------------------------------------------------------------------------

#define Ldep   6
#define Hdim   768
#define Edim   512
#define Bn     64
#define Sn     512
#define TMAX   12
#define Odim   1024

typedef __attribute__((ext_vector_type(16))) _Float16 v16h;
typedef __attribute__((ext_vector_type(8)))  float    v8f;

// ---------------- WMMA helpers (contiguous k order, same in A and B) ------------------------

// A-matrix 16x32 f16: lane L holds row M = L%16.
// lanes 0-15 : elems 0..15 = K(k0+0..15)
// lanes 16-31: elems 0..15 = K(k0+16..31)
__device__ __forceinline__ v16h load_a16x32(const float* __restrict__ A, int lda,
                                            int row, int k0, int lane) {
  const float* p = A + (size_t)row * lda + k0 + ((lane & 16) ? 16 : 0);
  v16h a;
#pragma unroll
  for (int j = 0; j < 8; ++j) {
    a[j]     = (_Float16)p[j];
    a[j + 8] = (_Float16)p[j + 8];
  }
  return a;
}

// Guarded scalar B-tile load (edge tiles only: N not multiple of 16).
__device__ __forceinline__ v16h load_b32x16(const float* __restrict__ B, int ldb,
                                            int n, int Ncols, int k0, int lane) {
  v16h b;
  if (n < Ncols) {
    const float* p = B + (size_t)(k0 + ((lane & 16) ? 16 : 0)) * ldb + n;
#pragma unroll
    for (int j = 0; j < 16; ++j) b[j] = (_Float16)p[(size_t)j * ldb];
  } else {
#pragma unroll
    for (int j = 0; j < 16; ++j) b[j] = (_Float16)0.0f;
  }
  return b;
}

// Read one lane's 32x16-tile column out of an LDS-staged fp32 tile.
// Tile layout: bs[k][n], 32 rows x 16 cols. lanes 0-15: K=0..15,
// lanes 16-31: K=16..31 (matches load_b32x16 order).
__device__ __forceinline__ v16h lds_b_column(const float* bs, int lane) {
  const float* p = bs + ((lane & 16) ? 16 * 16 : 0) + (lane & 15);
  v16h b;
#pragma unroll
  for (int j = 0; j < 16; ++j) b[j] = (_Float16)p[j * 16];
  return b;
}

__device__ __forceinline__ v8f wmma_f16(v16h a, v16h b, v8f c) {
  return __builtin_amdgcn_wmma_f32_16x16x32_f16(false, a, false, b,
                                                (short)0, c, false, false);
}

// Flat LDS addresses carry the aperture in addr[63:32]; addr[31:0] is the
// byte offset within LDS (ISA 10.2), so truncation yields the DS address.
__device__ __forceinline__ unsigned lds_off(const void* p) {
  return (unsigned)(size_t)p;
}

// ---------------- Generic WMMA GEMM: C = A[MxK] * B[KxN] + bias ------------
// grid = (ceil(N/16), ceil(M/16)), block = 32 (one wave). K multiple of 32.
__global__ __launch_bounds__(32)
void gemm_wmma(const float* __restrict__ A, int lda,
               const float* __restrict__ B, int ldb,
               const float* __restrict__ bias,
               float* __restrict__ C, int ldc, int coff,
               int M, int Ncols, int K, int relu,
               const int* __restrict__ seq, int t_guard, int m_from_seq) {
  if (seq) {
    int T = *seq; if (T > TMAX) T = TMAX;
    if (t_guard >= 0 && t_guard >= T) return;
    if (m_from_seq) M = Bn * T;
  }
  __shared__ float bs[32 * 16];                 // 2 KB fp32 B tile
  const int lane  = threadIdx.x;
  const int tileN = blockIdx.x, tileM = blockIdx.y;
  const int rowA  = tileM * 16 + (lane & 15);
  const int n     = tileN * 16 + (lane & 15);
  const bool full = (tileN * 16 + 16) <= Ncols; // interior tile: async path
  const unsigned dst = lds_off(bs) + (unsigned)lane * 64u;
  v8f c = {};
  for (int k0 = 0; k0 < K; k0 += 32) {
    v16h a = load_a16x32(A, lda, rowA, k0, lane);
    v16h b;
    if (full) {
      // Each lane DMAs one 64B row of the 32x16 fp32 tile into LDS.
      unsigned long long src =
          (unsigned long long)(B + (size_t)(k0 + lane) * ldb + tileN * 16);
      asm volatile(
          "s_wait_dscnt 0x0\n\t"                // prior-iter LDS reads done
          "global_load_async_to_lds_b128 %0, %1, off\n\t"
          "global_load_async_to_lds_b128 %0, %1, off offset:16\n\t"
          "global_load_async_to_lds_b128 %0, %1, off offset:32\n\t"
          "global_load_async_to_lds_b128 %0, %1, off offset:48\n\t"
          "s_wait_asynccnt 0x0"
          :: "v"(dst), "v"(src) : "memory");
      b = lds_b_column(bs, lane);
    } else {
      b = load_b32x16(B, ldb, n, Ncols, k0, lane);
    }
    c = wmma_f16(a, b, c);
  }
  const int colg = tileN * 16 + (lane & 15);
  float bv = (bias && colg < Ncols) ? bias[colg] : 0.0f;
#pragma unroll
  for (int r = 0; r < 8; ++r) {
    int rowg = tileM * 16 + r + ((lane & 16) ? 8 : 0);
    if (rowg < M && colg < Ncols) {
      float v = c[r] + bv;
      if (relu) v = fmaxf(v, 0.0f);
      C[(size_t)rowg * ldc + coff + colg] = v;
    }
  }
}

// ---------------- Fused RK4 stage: g = hh @ rec_W[768,2304] ---------------
// Wave computes 3 WMMA output tiles (cols n, n+H, n+2H = gates u,r,c), all
// B tiles async-DMA'd to LDS, then the LTC ODE update. grid=(48,4), block=32.
__global__ __launch_bounds__(32)
void rk4_stage(const float* __restrict__ HHin, const float* __restrict__ Hbase,
               const float* __restrict__ IC, const float* __restrict__ recW,
               const float* __restrict__ tau,
               float* __restrict__ Kout, float* __restrict__ HHnext,
               float coef, int final_stage,
               const float* __restrict__ K1, const float* __restrict__ K2,
               const float* __restrict__ K3,
               const int* __restrict__ seq, int t) {
  int T = *seq; if (T > TMAX) T = TMAX;
  if (t >= T) return;
  __shared__ float bsu[32 * 16];
  __shared__ float bsr[32 * 16];
  __shared__ float bsc[32 * 16];
  const int lane  = threadIdx.x;
  const int tileN = blockIdx.x, tileM = blockIdx.y;
  const int rowA  = tileM * 16 + (lane & 15);
  const int n0    = tileN * 16;
  const unsigned du = lds_off(bsu) + (unsigned)lane * 64u;
  const unsigned dr = lds_off(bsr) + (unsigned)lane * 64u;
  const unsigned dc = lds_off(bsc) + (unsigned)lane * 64u;
  v8f cu = {}, cr = {}, cc = {};
  for (int k0 = 0; k0 < Hdim; k0 += 32) {
    v16h a = load_a16x32(HHin, Hdim, rowA, k0, lane);
    const float* rrow = recW + (size_t)(k0 + lane) * (3 * Hdim);
    unsigned long long su = (unsigned long long)(rrow + n0);
    unsigned long long sr = (unsigned long long)(rrow + n0 + Hdim);
    unsigned long long sc = (unsigned long long)(rrow + n0 + 2 * Hdim);
    asm volatile(
        "s_wait_dscnt 0x0\n\t"
        "global_load_async_to_lds_b128 %0, %3, off\n\t"
        "global_load_async_to_lds_b128 %0, %3, off offset:16\n\t"
        "global_load_async_to_lds_b128 %0, %3, off offset:32\n\t"
        "global_load_async_to_lds_b128 %0, %3, off offset:48\n\t"
        "global_load_async_to_lds_b128 %1, %4, off\n\t"
        "global_load_async_to_lds_b128 %1, %4, off offset:16\n\t"
        "global_load_async_to_lds_b128 %1, %4, off offset:32\n\t"
        "global_load_async_to_lds_b128 %1, %4, off offset:48\n\t"
        "global_load_async_to_lds_b128 %2, %5, off\n\t"
        "global_load_async_to_lds_b128 %2, %5, off offset:16\n\t"
        "global_load_async_to_lds_b128 %2, %5, off offset:32\n\t"
        "global_load_async_to_lds_b128 %2, %5, off offset:48\n\t"
        "s_wait_asynccnt 0x0"
        :: "v"(du), "v"(dr), "v"(dc), "v"(su), "v"(sr), "v"(sc) : "memory");
    cu = wmma_f16(a, lds_b_column(bsu, lane), cu);
    cr = wmma_f16(a, lds_b_column(bsr, lane), cr);
    cc = wmma_f16(a, lds_b_column(bsc, lane), cc);
  }
  const int colg = n0 + (lane & 15);
  const float tj = fabsf(tau[colg]) + 1e-6f;
#pragma unroll
  for (int r = 0; r < 8; ++r) {
    int rowg = tileM * 16 + r + ((lane & 16) ? 8 : 0);
    size_t idx = (size_t)rowg * Hdim + colg;
    float hh = HHin[idx];
    float u  = 1.0f / (1.0f + __expf(-cu[r]));
    float rr = 1.0f / (1.0f + __expf(-cr[r]));
    float e2 = __expf(2.0f * cc[r]);
    float cv = (e2 - 1.0f) / (e2 + 1.0f);            // tanh
    float k  = 0.1f * (-hh * rr + (cv + IC[idx]) * u) / tj;
    Kout[idx] = k;
    if (final_stage)
      HHnext[idx] = Hbase[idx] +
                    (K1[idx] + 2.0f * K2[idx] + 2.0f * K3[idx] + k) * (1.0f / 6.0f);
    else
      HHnext[idx] = Hbase[idx] + coef * k;
  }
}

// ---------------- Row LayerNorm (optional residual / accumulate / relu) ----
// flags: bit0 = Y += LN, bit1 = relu, bit2 = rows limited to 64*seq.
__global__ __launch_bounds__(256)
void ln_kernel(const float* __restrict__ X, const float* __restrict__ Res,
               const float* __restrict__ g, const float* __restrict__ bb,
               float* __restrict__ Y, float* __restrict__ Y2,
               int ncols, int flags, const int* __restrict__ seq, int t) {
  int T = TMAX;
  if (seq) { T = *seq; if (T > TMAX) T = TMAX; if (t >= 0 && t >= T) return; }
  int row = blockIdx.x;
  if ((flags & 4) && row >= Bn * T) return;
  const float* xr = X + (size_t)row * ncols;
  const float* rr = Res ? Res + (size_t)row * ncols : nullptr;
  __shared__ float rowbuf[1024];
  __shared__ float red[256];
  int tid = threadIdx.x;
  float s1 = 0.0f;
  for (int c = tid; c < ncols; c += 256) {
    float v = xr[c] + (rr ? rr[c] : 0.0f);
    rowbuf[c] = v;
    s1 += v;
  }
  red[tid] = s1; __syncthreads();
  for (int sft = 128; sft > 0; sft >>= 1) {
    if (tid < sft) red[tid] += red[tid + sft];
    __syncthreads();
  }
  float mean = red[0] / (float)ncols;
  __syncthreads();
  float s2 = 0.0f;
  for (int c = tid; c < ncols; c += 256) { float d = rowbuf[c] - mean; s2 += d * d; }
  red[tid] = s2; __syncthreads();
  for (int sft = 128; sft > 0; sft >>= 1) {
    if (tid < sft) red[tid] += red[tid + sft];
    __syncthreads();
  }
  float inv = rsqrtf(red[0] / (float)ncols + 1e-5f);
  for (int c = tid; c < ncols; c += 256) {
    float v = (rowbuf[c] - mean) * inv * g[c] + bb[c];
    if (flags & 2) v = fmaxf(v, 0.0f);
    size_t idx = (size_t)row * ncols + c;
    if (flags & 1) Y[idx] += v; else Y[idx] = v;
    if (Y2) Y2[idx] = v;
  }
}

// ---------------- Embedding + PE + LN + mean over S ------------------------
// mean commutes with the downstream linear projection, so xp = (mean_s LN) @ W.
__global__ __launch_bounds__(256)
void embed_mean_kernel(const int* __restrict__ ids, const float* __restrict__ emb,
                       const float* __restrict__ g, const float* __restrict__ bb,
                       float* __restrict__ Y) {
  int b = blockIdx.x, tid = threadIdx.x;
  __shared__ float r1[256], r2[256];
  int c0 = tid, c1 = tid + 256;
  const float kf = 9.21034037198e0f / (float)Edim;   // ln(10000)/E
  float d0 = __expf(-(float)(c0 & ~1) * kf);
  float d1 = __expf(-(float)(c1 & ~1) * kf);
  float g0 = g[c0], g1 = g[c1], b0 = bb[c0], b1 = bb[c1];
  float acc0 = 0.0f, acc1 = 0.0f;
  for (int s = 0; s < Sn; ++s) {
    int id = ids[b * Sn + s];
    const float* er = emb + (size_t)id * Edim;
    float fs = (float)s;
    float v0 = er[c0] + ((c0 & 1) ? __cosf(fs * d0) : __sinf(fs * d0));
    float v1 = er[c1] + ((c1 & 1) ? __cosf(fs * d1) : __sinf(fs * d1));
    r1[tid] = v0 + v1;
    r2[tid] = v0 * v0 + v1 * v1;
    __syncthreads();
    for (int sft = 128; sft > 0; sft >>= 1) {
      if (tid < sft) { r1[tid] += r1[tid + sft]; r2[tid] += r2[tid + sft]; }
      __syncthreads();
    }
    float mean = r1[0] / (float)Edim;
    float var  = r2[0] / (float)Edim - mean * mean;
    float inv  = rsqrtf(var + 1e-5f);
    __syncthreads();
    acc0 += (v0 - mean) * inv * g0 + b0;
    acc1 += (v1 - mean) * inv * g1 + b1;
  }
  Y[(size_t)b * Edim + c0] = acc0 * (1.0f / (float)Sn);
  Y[(size_t)b * Edim + c1] = acc1 * (1.0f / (float)Sn);
}

// ---------------- Cell cross-attention core (4 heads, hd=192, Sq=1) --------
__global__ __launch_bounds__(32)
void cell_attn_kernel(const float* __restrict__ Q, const float* __restrict__ Kc,
                      const float* __restrict__ Vc, float* __restrict__ Ctx,
                      const int* __restrict__ seq, int t) {
  int T = *seq; if (T > TMAX) T = TMAX;
  if (t >= T) return;
  int nt = t; if (nt <= 0) return;
  int b = blockIdx.x >> 2, h = blockIdx.x & 3;
  int lane = threadIdx.x;
  int base = h * 192 + lane * 6;
  float qv[6];
#pragma unroll
  for (int j = 0; j < 6; ++j) qv[j] = Q[(size_t)b * Hdim + base + j];
  float s[TMAX];
  float mx = -3.0e38f;
  for (int tk = 0; tk < nt; ++tk) {
    const float* kr = Kc + ((size_t)tk * Bn + b) * Hdim + base;
    float d = 0.0f;
#pragma unroll
    for (int j = 0; j < 6; ++j) d += qv[j] * kr[j];
#pragma unroll
    for (int off = 16; off > 0; off >>= 1) d += __shfl_xor(d, off, 32);
    d *= 0.0721687836f;                     // 192^-0.5
    s[tk] = d; mx = fmaxf(mx, d);
  }
  float den = 0.0f;
  for (int tk = 0; tk < nt; ++tk) { s[tk] = __expf(s[tk] - mx); den += s[tk]; }
  float rden = 1.0f / den;
  float acc[6] = {0, 0, 0, 0, 0, 0};
  for (int tk = 0; tk < nt; ++tk) {
    const float* vr = Vc + ((size_t)tk * Bn + b) * Hdim + base;
    float w = s[tk] * rden;
#pragma unroll
    for (int j = 0; j < 6; ++j) acc[j] += w * vr[j];
  }
#pragma unroll
  for (int j = 0; j < 6; ++j) Ctx[(size_t)b * Hdim + base + j] = acc[j];
}

// ---------------- Global attention core (12 heads, hd=64, T x T) -----------
__global__ __launch_bounds__(32)
void glob_attn_kernel(const float* __restrict__ Qg, const float* __restrict__ Kg,
                      const float* __restrict__ Vg, float* __restrict__ Ctx,
                      const int* __restrict__ seq) {
  int T = *seq; if (T > TMAX) T = TMAX;
  int b = blockIdx.x / 12, h = blockIdx.x % 12;
  int lane = threadIdx.x;
  int base = h * 64 + lane * 2;
  for (int tq = 0; tq < T; ++tq) {
    const float* qr = Qg + ((size_t)tq * Bn + b) * Hdim + base;
    float q0 = qr[0], q1 = qr[1];
    float s[TMAX]; float mx = -3.0e38f;
    for (int tk = 0; tk < T; ++tk) {
      const float* kr = Kg + ((size_t)tk * Bn + b) * Hdim + base;
      float d = q0 * kr[0] + q1 * kr[1];
#pragma unroll
      for (int off = 16; off > 0; off >>= 1) d += __shfl_xor(d, off, 32);
      d *= 0.125f;                          // 64^-0.5
      s[tk] = d; mx = fmaxf(mx, d);
    }
    float den = 0.0f;
    for (int tk = 0; tk < T; ++tk) { s[tk] = __expf(s[tk] - mx); den += s[tk]; }
    float rden = 1.0f / den;
    float a0v = 0.0f, a1v = 0.0f;
    for (int tk = 0; tk < T; ++tk) {
      const float* vr = Vg + ((size_t)tk * Bn + b) * Hdim + base;
      float w = s[tk] * rden;
      a0v += w * vr[0]; a1v += w * vr[1];
    }
    float* o = Ctx + ((size_t)tq * Bn + b) * Hdim + base;
    o[0] = a0v; o[1] = a1v;
  }
}

// ---------------- Mean-pool over time --------------------------------------
__global__ __launch_bounds__(256)
void pool_mean_kernel(const float* __restrict__ hist2, float* __restrict__ pooled,
                      const int* __restrict__ seq) {
  int T = *seq; if (T > TMAX) T = TMAX; if (T < 1) T = 1;
  int b = blockIdx.x;
  for (int j = threadIdx.x; j < Hdim; j += 256) {
    float a = 0.0f;
    for (int t = 0; t < T; ++t) a += hist2[((size_t)t * Bn + b) * Hdim + j];
    pooled[(size_t)b * Hdim + j] = a / (float)T;
  }
}

__global__ __launch_bounds__(256)
void zero_kernel(float* __restrict__ p, int n) {
  int i = blockIdx.x * 256 + threadIdx.x;
  if (i < n) p[i] = 0.0f;
}

// ---------------------------------------------------------------------------
extern "C" void kernel_launch(void* const* d_in, const int* in_sizes, int n_in,
                              void* d_out, int out_size, void* d_ws, size_t ws_size,
                              hipStream_t stream) {
  (void)in_sizes; (void)n_in; (void)out_size; (void)ws_size;
  // Input order: input_ids, then params dict in insertion order, then seq_len.
  enum {
    i_ids = 0, p_emb, p_embed_ln_g, p_embed_ln_b, p_proj_W, p_proj_b,
    p_tau, p_in_W, p_in_b, p_in_ln_g, p_in_ln_b, p_rec_W,
    p_cWq, p_cbq, p_cWk, p_cbk, p_cWv, p_cbv, p_cWo, p_cbo,
    p_attn_ln_g, p_attn_ln_b, p_lyr_ln_g, p_lyr_ln_b,
    p_gWq, p_gbq, p_gWk, p_gbk, p_gWv, p_gbv, p_gWo, p_gbo,
    p_glob_ln_g, p_glob_ln_b,
    p_op1_W, p_op1_b, p_op_ln1_g, p_op_ln1_b,
    p_op2_W, p_op2_b, p_op_ln2_g, p_op_ln2_b,
    p_acc1_W, p_acc1_b, p_acc2_W, p_acc2_b,
    p_met1_W, p_met1_b, p_met2_W, p_met2_b,
    p_exp1_W, p_exp1_b, p_exp2_W, p_exp2_b,
    p_conf1_W, p_conf1_b, p_conf2_W, p_conf2_b,
    i_seq
  };
  auto P = [&](int i) { return (const float*)d_in[i]; };
  const int*   ids  = (const int*)d_in[i_ids];
  const int*   seqp = (const int*)d_in[i_seq];
  float*       out  = (float*)d_out;

  // -------- workspace carve-up (floats) --------
  float* W = (float*)d_ws;
  size_t o = 0;
  auto alloc = [&](size_t n) { float* p = W + o; o += n; return p; };
  float* ymean = alloc((size_t)Bn * Edim);
  float* xp    = alloc((size_t)Bn * Hdim);
  float* hid   = alloc((size_t)Ldep * Bn * Hdim);
  float* hist  = alloc((size_t)TMAX * Bn * Hdim);              // time-major
  float* kc    = alloc((size_t)Ldep * TMAX * Bn * Hdim);
  float* vc    = alloc((size_t)Ldep * TMAX * Bn * Hdim);
  float* ic    = alloc((size_t)Bn * Hdim);
  float* icraw = alloc((size_t)Bn * Hdim);
  float* qb    = alloc((size_t)Bn * Hdim);
  float* ctx   = alloc((size_t)Bn * Hdim);
  float* a0    = alloc((size_t)Bn * Hdim);
  float* k1    = alloc((size_t)Bn * Hdim);
  float* k2    = alloc((size_t)Bn * Hdim);
  float* k3    = alloc((size_t)Bn * Hdim);
  float* k4    = alloc((size_t)Bn * Hdim);
  float* hta   = alloc((size_t)Bn * Hdim);
  float* htb   = alloc((size_t)Bn * Hdim);
  float* hnew  = alloc((size_t)Bn * Hdim);
  float* Qg    = alloc((size_t)TMAX * Bn * Hdim);
  float* Kg    = alloc((size_t)TMAX * Bn * Hdim);
  float* Vg    = alloc((size_t)TMAX * Bn * Hdim);
  float* ctxg  = alloc((size_t)TMAX * Bn * Hdim);
  float* attng = alloc((size_t)TMAX * Bn * Hdim);
  float* hist2 = alloc((size_t)TMAX * Bn * Hdim);
  float* pooled= alloc((size_t)Bn * Hdim);
  float* f1raw = alloc((size_t)Bn * Odim);
  float* f1    = alloc((size_t)Bn * Odim);
  float* f2raw = alloc((size_t)Bn * Odim);
  float* f2    = alloc((size_t)Bn * Odim);
  float* hacc  = alloc((size_t)Bn * 256);
  float* hconf = alloc((size_t)Bn * 128);

  auto gemm = [&](const float* A, int lda, const float* Bm, int ldb,
                  const float* bias, float* C, int ldc, int coff,
                  int M, int N, int K, int relu,
                  const int* seq, int tg, int mseq) {
    dim3 g((N + 15) / 16, (M + 15) / 16);
    gemm_wmma<<<g, dim3(32), 0, stream>>>(A, lda, Bm, ldb, bias, C, ldc, coff,
                                          M, N, K, relu, seq, tg, mseq);
  };
  const size_t HH = (size_t)Hdim * Hdim;

  // -------- prologue: embed + mean, then xp = y @ proj_W + proj_b ----------
  {
    int nz = Ldep * Bn * Hdim;
    zero_kernel<<<(nz + 255) / 256, 256, 0, stream>>>(hid, nz);
  }
  embed_mean_kernel<<<Bn, 256, 0, stream>>>(ids, P(p_emb), P(p_embed_ln_g),
                                            P(p_embed_ln_b), ymean);
  gemm(ymean, Edim, P(p_proj_W), Hdim, P(p_proj_b), xp, Hdim, 0,
       Bn, Hdim, Edim, 0, nullptr, -1, 0);

  // -------- sequential LTC timestep loop -----------------------------------
  for (int t = 0; t < TMAX; ++t) {
    if (t >= 1) {
      const float* hrow = hist + (size_t)(t - 1) * Bn * Hdim;
      for (int i = 0; i < Ldep; ++i) {
        float* kdst = kc + ((size_t)i * TMAX + (t - 1)) * Bn * Hdim;
        float* vdst = vc + ((size_t)i * TMAX + (t - 1)) * Bn * Hdim;
        gemm(hrow, Hdim, P(p_cWk) + (size_t)i * HH, Hdim, P(p_cbk) + i * Hdim,
             kdst, Hdim, 0, Bn, Hdim, Hdim, 0, seqp, t, 0);
        gemm(hrow, Hdim, P(p_cWv) + (size_t)i * HH, Hdim, P(p_cbv) + i * Hdim,
             vdst, Hdim, 0, Bn, Hdim, Hdim, 0, seqp, t, 0);
      }
    }
    for (int i = 0; i < Ldep; ++i) {
      const float* li = (i == 0) ? xp : hid + (size_t)(i - 1) * Bn * Hdim;
      float* hi = hid + (size_t)i * Bn * Hdim;
      // ic = LN(li @ in_W + in_b)
      gemm(li, Hdim, P(p_in_W) + (size_t)i * HH, Hdim, P(p_in_b) + i * Hdim,
           icraw, Hdim, 0, Bn, Hdim, Hdim, 0, seqp, t, 0);
      ln_kernel<<<Bn, 256, 0, stream>>>(icraw, nullptr, P(p_in_ln_g) + i * Hdim,
                                        P(p_in_ln_b) + i * Hdim, ic, nullptr,
                                        Hdim, 0, seqp, t);
      if (t >= 1) {
        gemm(hi, Hdim, P(p_cWq) + (size_t)i * HH, Hdim, P(p_cbq) + i * Hdim,
             qb, Hdim, 0, Bn, Hdim, Hdim, 0, seqp, t, 0);
        cell_attn_kernel<<<Bn * 4, 32, 0, stream>>>(
            qb, kc + (size_t)i * TMAX * Bn * Hdim,
            vc + (size_t)i * TMAX * Bn * Hdim, ctx, seqp, t);
        gemm(ctx, Hdim, P(p_cWo) + (size_t)i * HH, Hdim, P(p_cbo) + i * Hdim,
             a0, Hdim, 0, Bn, Hdim, Hdim, 0, seqp, t, 0);
        // ic += LN(attn_out)
        ln_kernel<<<Bn, 256, 0, stream>>>(a0, nullptr, P(p_attn_ln_g) + i * Hdim,
                                          P(p_attn_ln_b) + i * Hdim, ic, nullptr,
                                          Hdim, 1, seqp, t);
      }
      // RK4: four fused GEMM+ODE stages
      const float* rw = P(p_rec_W) + (size_t)i * Hdim * 3 * Hdim;
      const float* tp = P(p_tau) + i * Hdim;
      dim3 gs(Hdim / 16, Bn / 16);
      rk4_stage<<<gs, 32, 0, stream>>>(hi,  hi, ic, rw, tp, k1, hta, 0.5f, 0,
                                       nullptr, nullptr, nullptr, seqp, t);
      rk4_stage<<<gs, 32, 0, stream>>>(hta, hi, ic, rw, tp, k2, htb, 0.5f, 0,
                                       nullptr, nullptr, nullptr, seqp, t);
      rk4_stage<<<gs, 32, 0, stream>>>(htb, hi, ic, rw, tp, k3, hta, 1.0f, 0,
                                       nullptr, nullptr, nullptr, seqp, t);
      rk4_stage<<<gs, 32, 0, stream>>>(hta, hi, ic, rw, tp, k4, hnew, 0.0f, 1,
                                       k1, k2, k3, seqp, t);
      // hiddens[i] = LN(h_new + li); last layer also appended to history
      float* y2 = (i == Ldep - 1) ? hist + (size_t)t * Bn * Hdim : nullptr;
      ln_kernel<<<Bn, 256, 0, stream>>>(hnew, li, P(p_lyr_ln_g) + i * Hdim,
                                        P(p_lyr_ln_b) + i * Hdim, hi, y2,
                                        Hdim, 0, seqp, t);
    }
  }

  // -------- global MHA over history (12 heads), rows = 64*T ---------------
  gemm(hist, Hdim, P(p_gWq), Hdim, P(p_gbq), Qg, Hdim, 0,
       TMAX * Bn, Hdim, Hdim, 0, seqp, -1, 1);
  gemm(hist, Hdim, P(p_gWk), Hdim, P(p_gbk), Kg, Hdim, 0,
       TMAX * Bn, Hdim, Hdim, 0, seqp, -1, 1);
  gemm(hist, Hdim, P(p_gWv), Hdim, P(p_gbv), Vg, Hdim, 0,
       TMAX * Bn, Hdim, Hdim, 0, seqp, -1, 1);
  glob_attn_kernel<<<Bn * 12, 32, 0, stream>>>(Qg, Kg, Vg, ctxg, seqp);
  gemm(ctxg, Hdim, P(p_gWo), Hdim, P(p_gbo), attng, Hdim, 0,
       TMAX * Bn, Hdim, Hdim, 0, seqp, -1, 1);
  ln_kernel<<<TMAX * Bn, 256, 0, stream>>>(attng, hist, P(p_glob_ln_g),
                                           P(p_glob_ln_b), hist2, nullptr,
                                           Hdim, 4, seqp, -1);
  pool_mean_kernel<<<Bn, 256, 0, stream>>>(hist2, pooled, seqp);

  // -------- readout MLP ----------------------------------------------------
  gemm(pooled, Hdim, P(p_op1_W), Odim, P(p_op1_b), f1raw, Odim, 0,
       Bn, Odim, Hdim, 0, nullptr, -1, 0);
  ln_kernel<<<Bn, 256, 0, stream>>>(f1raw, nullptr, P(p_op_ln1_g), P(p_op_ln1_b),
                                    f1, nullptr, Odim, 2, nullptr, -1);
  gemm(f1, Odim, P(p_op2_W), Odim, P(p_op2_b), f2raw, Odim, 0,
       Bn, Odim, Odim, 0, nullptr, -1, 0);
  ln_kernel<<<Bn, 256, 0, stream>>>(f2raw, nullptr, P(p_op_ln2_g), P(p_op_ln2_b),
                                    f2, nullptr, Odim, 2, nullptr, -1);

  // -------- heads -> d_out [64, 304] = [acc(101) | met(101) | exp(101) | conf(1)]
  const int OLD = 304;
  gemm(f2, Odim, P(p_acc1_W), 256, P(p_acc1_b), hacc, 256, 0,
       Bn, 256, Odim, 1, nullptr, -1, 0);
  gemm(hacc, 256, P(p_acc2_W), 101, P(p_acc2_b), out, OLD, 0,
       Bn, 101, 256, 0, nullptr, -1, 0);
  gemm(f2, Odim, P(p_met1_W), 256, P(p_met1_b), hacc, 256, 0,
       Bn, 256, Odim, 1, nullptr, -1, 0);
  gemm(hacc, 256, P(p_met2_W), 101, P(p_met2_b), out, OLD, 101,
       Bn, 101, 256, 0, nullptr, -1, 0);
  gemm(f2, Odim, P(p_exp1_W), 256, P(p_exp1_b), hacc, 256, 0,
       Bn, 256, Odim, 1, nullptr, -1, 0);
  gemm(hacc, 256, P(p_exp2_W), 101, P(p_exp2_b), out, OLD, 202,
       Bn, 101, 256, 0, nullptr, -1, 0);
  gemm(f2, Odim, P(p_conf1_W), 128, P(p_conf1_b), hconf, 128, 0,
       Bn, 128, Odim, 1, nullptr, -1, 0);
  gemm(hconf, 128, P(p_conf2_W), 1, P(p_conf2_b), out, OLD, 303,
       Bn, 1, 128, 0, nullptr, -1, 0);
}
